// MambaLayer_13348758356004
// MI455X (gfx1250) — matrix operand order repaired and checked by reading, not compiled, not run on hardware
//
#include <hip/hip_runtime.h>
#include <hip/hip_bf16.h>
#include <math.h>

// ---------------- problem constants (from reference) ----------------
#define DIM      1024
#define D_STATE  16
#define D_CONV   4
#define D_INNER  2048            // EXPAND * DIM
#define DT_RANK  64              // DIM / 16
#define BATCH    4096
#define XPROJ_N  (DT_RANK + 2 * D_STATE)   // 96
#define XPROJ_NP 128                       // padded to a multiple of 64
#define LN_EPS   1e-5f

typedef __bf16 bf16;
typedef __attribute__((ext_vector_type(16))) bf16  v16bf;
typedef __attribute__((ext_vector_type(8)))  bf16  v8bf;
typedef __attribute__((ext_vector_type(8)))  float v8f;

union AFrag { v16bf v; v8bf h[2]; };

__device__ __forceinline__ float silu_f(float x) { return x / (1.0f + __expf(-x)); }
__device__ __forceinline__ float softplus_f(float x) {
    return (x > 20.0f) ? x : log1pf(__expf(x));
}

// ---------------------------------------------------------------
// bf16 WMMA GEMM:  C[M,N] = A[M,K] * B[N,K]^T  (fp32 out)
// A: row-major (lda elems). B: row-major, N rows of K (ldb elems).
// Each wave computes a 16(M) x 64(N) strip: 1 A-frag feeds 4 WMMAs.
// Software-pipelined: iteration i+1's 10 loads are issued into a
// second register set before iteration i's WMMAs, so the compiler
// can emit partial s_wait_loadcnt and overlap loads with WMMA.
// REQUIREMENTS: M % 16 == 0, N % 64 == 0, K % 32 == 0  (caller pads).
// Wave index is readfirstlane'd so all tile math is scalar -> no
// EXEC manipulation around WMMA (ISA 7.12 requires EXEC all-ones).
// A/B fragments use contiguous k = 16*hi + s (wave32), the same order in both.
// ---------------------------------------------------------------
__global__ void __launch_bounds__(256)
gemm_bf16_wmma(const bf16* __restrict__ A, const bf16* __restrict__ B,
               float* __restrict__ C, int M, int N, int K,
               int lda, int ldb, int ldc)
{
    const int lane  = threadIdx.x & 31;
    // force the wave id into an SGPR: uniform by construction
    const int winb  = __builtin_amdgcn_readfirstlane(threadIdx.x >> 5);
    const int wave  = blockIdx.x * 8 + winb;           // 8 waves / 256-thr block
    const int tiles_m  = M >> 4;
    const int tiles_n4 = N >> 6;
    if (wave >= tiles_m * tiles_n4) return;            // scalar branch

    const int tm   = wave % tiles_m;
    const int tn   = wave / tiles_m;
    const int row0 = tm << 4;
    const int col0 = tn << 6;

    const int hi   = lane >> 4;          // lane half: 0 (lanes 0-15) / 1 (16-31)
    const int mrow = row0 + (lane & 15); // A-frag: this lane's matrix row
    const int ncol = lane & 15;          // B-frag: this lane's matrix column
    const int kA0  = hi * 16;            // A elems 0..7  hold K = k0+kA0 .. +7
    const int kA1  = kA0 + 8;            // A elems 8..15 hold K = k0+kA1 .. +7
    const int kB   = hi * 16;            // B elems 0..15 hold K = k0+kB  .. +15

    v8f acc0 = {}, acc1 = {}, acc2 = {}, acc3 = {};

    const bf16* __restrict__ Arow = A + (size_t)mrow * lda;
    const bf16* __restrict__ B0 = B + (size_t)(col0 +  0 + ncol) * ldb + kB;
    const bf16* __restrict__ B1 = B + (size_t)(col0 + 16 + ncol) * ldb + kB;
    const bf16* __restrict__ B2 = B + (size_t)(col0 + 32 + ncol) * ldb + kB;
    const bf16* __restrict__ B3 = B + (size_t)(col0 + 48 + ncol) * ldb + kB;

    // ---- pipeline prologue: load k0 = 0 fragments ----
    AFrag a;
    a.h[0] = *(const v8bf*)(Arow + kA0);
    a.h[1] = *(const v8bf*)(Arow + kA1);
    v16bf b0 = *(const v16bf*)(B0);
    v16bf b1 = *(const v16bf*)(B1);
    v16bf b2 = *(const v16bf*)(B2);
    v16bf b3 = *(const v16bf*)(B3);

    const int Klast = K - 32;
    for (int k0 = 0; k0 < K; k0 += 32) {
        // next-iteration K offset (clamped: last iter harmlessly reloads)
        const int kn = (k0 < Klast) ? (k0 + 32) : k0;

        // speculative prefetch two iterations ahead (global_prefetch_b8)
        __builtin_prefetch(Arow + kn + 32, 0, 3);

        // issue ALL next-iteration loads before this iteration's WMMAs
        AFrag an;
        an.h[0] = *(const v8bf*)(Arow + kn + kA0);
        an.h[1] = *(const v8bf*)(Arow + kn + kA1);
        v16bf n0 = *(const v16bf*)(B0 + kn);
        v16bf n1 = *(const v16bf*)(B1 + kn);
        v16bf n2 = *(const v16bf*)(B2 + kn);
        v16bf n3 = *(const v16bf*)(B3 + kn);

        acc0 = __builtin_amdgcn_wmma_f32_16x16x32_bf16(false, a.v, false, b0,
                                                       (short)0, acc0, false, false);
        acc1 = __builtin_amdgcn_wmma_f32_16x16x32_bf16(false, a.v, false, b1,
                                                       (short)0, acc1, false, false);
        acc2 = __builtin_amdgcn_wmma_f32_16x16x32_bf16(false, a.v, false, b2,
                                                       (short)0, acc2, false, false);
        acc3 = __builtin_amdgcn_wmma_f32_16x16x32_bf16(false, a.v, false, b3,
                                                       (short)0, acc3, false, false);

        a = an; b0 = n0; b1 = n1; b2 = n2; b3 = n3;
    }

    // C/D layout: VGPR r, lane l -> row = row0 + r + 8*hi, col = col0+16t+(l&15)
    float* Crow = C + (size_t)(row0 + hi * 8) * ldc + col0 + ncol;
#pragma unroll
    for (int r = 0; r < 8; ++r) {
        float* Cp = Crow + (size_t)r * ldc;
        Cp[0]  = acc0[r];
        Cp[16] = acc1[r];
        Cp[32] = acc2[r];
        Cp[48] = acc3[r];
    }
}

// --------------- fp32 -> bf16 conversion (weights) ---------------
__global__ void cvt_f32_bf16(const float* __restrict__ src, bf16* __restrict__ dst, int n)
{
    int i = blockIdx.x * blockDim.x + threadIdx.x;
    int stride = gridDim.x * blockDim.x;
    for (; i < n; i += stride) dst[i] = (bf16)src[i];
}

// --------------- bf16 zero fill (pad rows) -----------------------
__global__ void fill0_bf16(bf16* __restrict__ dst, int n)
{
    int i = blockIdx.x * blockDim.x + threadIdx.x;
    int stride = gridDim.x * blockDim.x;
    for (; i < n; i += stride) dst[i] = (bf16)0.0f;
}

// --------------- LayerNorm over DIM, emit bf16 -------------------
__global__ void __launch_bounds__(256)
ln_kernel(const float* __restrict__ x, const float* __restrict__ gamma,
          const float* __restrict__ beta, bf16* __restrict__ xn)
{
    __shared__ float sh1[256];
    __shared__ float sh2[256];
    const int b = blockIdx.x;
    const int t = threadIdx.x;
    const float* xr = x + (size_t)b * DIM;

    float s = 0.f, s2 = 0.f;
    for (int i = t; i < DIM; i += 256) { float v = xr[i]; s += v; s2 += v * v; }
    sh1[t] = s; sh2[t] = s2;
    __syncthreads();
    for (int off = 128; off > 0; off >>= 1) {
        if (t < off) { sh1[t] += sh1[t + off]; sh2[t] += sh2[t + off]; }
        __syncthreads();
    }
    const float mu  = sh1[0] * (1.0f / DIM);
    const float var = sh2[0] * (1.0f / DIM) - mu * mu;
    const float rs  = rsqrtf(var + LN_EPS);
    for (int i = t; i < DIM; i += 256) {
        float v = (xr[i] - mu) * rs * gamma[i] + beta[i];
        xn[(size_t)b * DIM + i] = (bf16)v;
    }
}

// ---- epi1: split xz, degenerate conv (only tap 3 hits data), silu ----
// xc = silu(conv_b[e] + conv_w[e,3]*xi[b,e]);  zs = silu(z[b,e])
__global__ void epi1_kernel(const float* __restrict__ xz,
                            const float* __restrict__ conv_w,
                            const float* __restrict__ conv_b,
                            float* __restrict__ xc_f, bf16* __restrict__ xc_b,
                            float* __restrict__ zs)
{
    int idx = blockIdx.x * blockDim.x + threadIdx.x;
    const int n = BATCH * D_INNER;
    if (idx >= n) return;
    const int b = idx / D_INNER;
    const int e = idx - b * D_INNER;
    const float xi = xz[(size_t)b * (2 * D_INNER) + e];
    const float z  = xz[(size_t)b * (2 * D_INNER) + D_INNER + e];
    const float pre = conv_b[e] + conv_w[e * D_CONV + (D_CONV - 1)] * xi;
    const float xc  = silu_f(pre);
    xc_f[idx] = xc;
    xc_b[idx] = (bf16)xc;
    zs[idx]   = silu_f(z);
}

// ---- epi2: per-row  s[b] = dot(Bm, Cm);  dtA_bf = bf16(xdb[:, :DT_RANK]) ----
__global__ void epi2_kernel(const float* __restrict__ xdb,   // ld = XPROJ_NP
                            float* __restrict__ s_arr, bf16* __restrict__ dtA)
{
    int b = blockIdx.x * blockDim.x + threadIdx.x;
    if (b >= BATCH) return;
    const float* r = xdb + (size_t)b * XPROJ_NP;
    float s = 0.f;
#pragma unroll
    for (int n = 0; n < D_STATE; ++n)
        s += r[DT_RANK + n] * r[DT_RANK + D_STATE + n];
    s_arr[b] = s;
#pragma unroll
    for (int k = 0; k < DT_RANK; ++k)
        dtA[(size_t)b * DT_RANK + k] = (bf16)r[k];
}

// ---- epi3: y = (softplus(dtp + b_dt)*s + Dskip) * xc * silu(z) ----
__global__ void epi3_kernel(const float* __restrict__ dtp,
                            const float* __restrict__ b_dt,
                            const float* __restrict__ Dskip,
                            const float* __restrict__ s_arr,
                            const float* __restrict__ xc_f,
                            const float* __restrict__ zs,
                            bf16* __restrict__ y_b)
{
    int idx = blockIdx.x * blockDim.x + threadIdx.x;
    const int n = BATCH * D_INNER;
    if (idx >= n) return;
    const int b = idx / D_INNER;
    const int d = idx - b * D_INNER;
    const float dt = softplus_f(dtp[idx] + b_dt[d]);
    const float y  = (dt * s_arr[b] + Dskip[d]) * xc_f[idx] * zs[idx];
    y_b[idx] = (bf16)y;
}

// ------------------------- launcher -------------------------
static inline int gemm_blocks(int M, int N) {
    int waves = (M >> 4) * (N >> 6);
    return (waves + 7) / 8;   // 8 waves per 256-thread block
}

extern "C" void kernel_launch(void* const* d_in, const int* in_sizes, int n_in,
                              void* d_out, int out_size, void* d_ws, size_t ws_size,
                              hipStream_t stream)
{
    (void)in_sizes; (void)n_in; (void)out_size; (void)ws_size;

    const float* x        = (const float*)d_in[0];
    const float* ln_gamma = (const float*)d_in[1];
    const float* ln_beta  = (const float*)d_in[2];
    const float* W_in     = (const float*)d_in[3];   // (4096, 1024)
    const float* conv_w   = (const float*)d_in[4];   // (2048, 4)
    const float* conv_b   = (const float*)d_in[5];   // (2048,)
    const float* W_xproj  = (const float*)d_in[6];   // (96, 2048)
    const float* W_dt     = (const float*)d_in[7];   // (2048, 64)
    const float* b_dt     = (const float*)d_in[8];   // (2048,)
    /* d_in[9] = A_log: provably unused (h0 == 0, single scan step) */
    const float* Dskip    = (const float*)d_in[10];  // (2048,)
    const float* W_out    = (const float*)d_in[11];  // (1024, 2048)
    float* out = (float*)d_out;

    // ---- workspace layout (bytes) ----
    char* ws = (char*)d_ws;
    size_t off = 0;
    auto alloc = [&](size_t bytes) { void* p = ws + off; off += (bytes + 255) & ~(size_t)255; return p; };
    bf16*  w_in_b   = (bf16*) alloc((size_t)2 * D_INNER * DIM * 2);        // 8 MB
    bf16*  w_out_b  = (bf16*) alloc((size_t)DIM * D_INNER * 2);            // 4 MB
    bf16*  w_xp_b   = (bf16*) alloc((size_t)XPROJ_NP * D_INNER * 2);       // 512 KB (padded)
    bf16*  w_dt_b   = (bf16*) alloc((size_t)D_INNER * DT_RANK * 2);        // 256 KB
    bf16*  xn_b     = (bf16*) alloc((size_t)BATCH * DIM * 2);              // 8 MB
    float* xz_f     = (float*)alloc((size_t)BATCH * 2 * D_INNER * 4);      // 64 MB
    float* xc_f     = (float*)alloc((size_t)BATCH * D_INNER * 4);          // 32 MB
    bf16*  xc_b     = (bf16*) alloc((size_t)BATCH * D_INNER * 2);          // 16 MB
    float* zs_f     = (float*)alloc((size_t)BATCH * D_INNER * 4);          // 32 MB
    float* xdb_f    = (float*)alloc((size_t)BATCH * XPROJ_NP * 4);         // 2 MB (padded)
    bf16*  dtA_b    = (bf16*) alloc((size_t)BATCH * DT_RANK * 2);          // 512 KB
    float* s_f      = (float*)alloc((size_t)BATCH * 4);                    // 16 KB
    float* dtp_f    = (float*)alloc((size_t)BATCH * D_INNER * 4);          // 32 MB
    bf16*  y_b      = (bf16*) alloc((size_t)BATCH * D_INNER * 2);          // 16 MB

    // ---- weight fp32 -> bf16 converts (memory-trivial vs. 23.3 TB/s) ----
    cvt_f32_bf16<<<1024, 256, 0, stream>>>(W_in,    w_in_b,  2 * D_INNER * DIM);
    cvt_f32_bf16<<<1024, 256, 0, stream>>>(W_out,   w_out_b, DIM * D_INNER);
    cvt_f32_bf16<<<256,  256, 0, stream>>>(W_xproj, w_xp_b,  XPROJ_N * D_INNER);
    // zero the 32 pad rows so padded GEMM columns are well-defined
    fill0_bf16<<<256, 256, 0, stream>>>(w_xp_b + (size_t)XPROJ_N * D_INNER,
                                        (XPROJ_NP - XPROJ_N) * D_INNER);
    cvt_f32_bf16<<<256,  256, 0, stream>>>(W_dt,    w_dt_b,  D_INNER * DT_RANK);

    // ---- 1. LayerNorm -> xn (bf16) ----
    ln_kernel<<<BATCH, 256, 0, stream>>>(x, ln_gamma, ln_beta, xn_b);

    // ---- 2. xz = xn @ W_in^T   (M=4096, N=4096, K=1024) ----
    gemm_bf16_wmma<<<gemm_blocks(BATCH, 2 * D_INNER), 256, 0, stream>>>(
        xn_b, w_in_b, xz_f, BATCH, 2 * D_INNER, DIM, DIM, DIM, 2 * D_INNER);

    // ---- 3. degenerate conv + silu, silu(z) ----
    {
        int n = BATCH * D_INNER;
        epi1_kernel<<<(n + 255) / 256, 256, 0, stream>>>(xz_f, conv_w, conv_b,
                                                         xc_f, xc_b, zs_f);
    }

    // ---- 4. xdb = xc @ W_xproj^T   (M=4096, N=128 padded, K=2048) ----
    gemm_bf16_wmma<<<gemm_blocks(BATCH, XPROJ_NP), 256, 0, stream>>>(
        xc_b, w_xp_b, xdb_f, BATCH, XPROJ_NP, D_INNER, D_INNER, D_INNER, XPROJ_NP);

    // ---- 5. s[b] = Bm . Cm ; dt_low -> bf16 ----
    epi2_kernel<<<(BATCH + 255) / 256, 256, 0, stream>>>(xdb_f, s_f, dtA_b);

    // ---- 6. dtp = dt_low @ W_dt^T   (M=4096, N=2048, K=64) ----
    gemm_bf16_wmma<<<gemm_blocks(BATCH, D_INNER), 256, 0, stream>>>(
        dtA_b, w_dt_b, dtp_f, BATCH, D_INNER, DT_RANK, DT_RANK, DT_RANK, D_INNER);

    // ---- 7. y = (softplus(dtp+b_dt)*s + Dskip) * xc * silu(z) ----
    {
        int n = BATCH * D_INNER;
        epi3_kernel<<<(n + 255) / 256, 256, 0, stream>>>(dtp_f, b_dt, Dskip, s_f,
                                                         xc_f, zs_f, y_b);
    }

    // ---- 8. out = y @ W_out^T   (M=4096, N=1024, K=2048) ----
    gemm_bf16_wmma<<<gemm_blocks(BATCH, DIM), 256, 0, stream>>>(
        y_b, w_out_b, out, BATCH, DIM, D_INNER, D_INNER, D_INNER, DIM);
}
